// GATLayer_42674795053752
// MI455X (gfx1250) — matrix operand order repaired and checked by reading, not compiled, not run on hardware
//
#include <hip/hip_runtime.h>
#include <hip/hip_bf16.h>

// GAT layer: B=4, N=2048, FIN=256, FOUT=256, H=4, D=64
// out = concat(h_new [B,N,FOUT], alpha_mean [B,N,N]) as fp32.

typedef __attribute__((ext_vector_type(2)))  float     v2f;
typedef __attribute__((ext_vector_type(8)))  float     v8f;
typedef __attribute__((ext_vector_type(2)))  _Float16  v2h;
typedef __attribute__((ext_vector_type(8)))  _Float16  v8h;
typedef __attribute__((ext_vector_type(16))) _Float16  v16h;
typedef __attribute__((address_space(3)))    float     lds_f;

constexpr int Bc   = 4;
constexpr int Nc   = 2048;
constexpr int FINc = 256;
constexpr int FOUTc= 256;
constexpr int Hc   = 4;
constexpr int Dc   = 64;

__device__ __forceinline__ float lrelu(float x) { return x > 0.0f ? x : 0.2f * x; }

// ---------------------------------------------------------------------------
// Kernel 1: hp = h @ W.T  (V_WMMA_F32_16X16X4_F32, exact fp32)
// Writes hp   in (B,H,N,D) fp32   (for s/t and softmax stats)
// Writes hpT16 in (B,H,D,N) f16   (transposed: B-fragment feed for PV WMMA)
// One wave per 16-row x 64-col (one head) tile.
// ---------------------------------------------------------------------------
__global__ __launch_bounds__(32) void k_project(const float* __restrict__ h,
                                                const float* __restrict__ W,
                                                float* __restrict__ hp,
                                                _Float16* __restrict__ hpT16) {
  const int lane = threadIdx.x;
  const int iLoc = lane & 15;
  const int half = lane >> 4;          // K offset 0 / 2
  const int r0   = blockIdx.x * 16;    // flattened row in [0, B*N)
  const int b    = r0 >> 11;
  const int n0   = r0 & (Nc - 1);
  const int head = blockIdx.y;
  const int c0   = head * Dc;

  v8f acc[4] = {};

  const float* aRow = h + (size_t)(r0 + iLoc) * FINc;
  for (int k0 = 0; k0 < FINc; k0 += 4) {
    const int kk = k0 + 2 * half;
    v2f a = *(const v2f*)(aRow + kk);
#pragma unroll
    for (int dt = 0; dt < 4; ++dt) {
      const int c = c0 + dt * 16 + iLoc;
      v2f bf = *(const v2f*)(W + (size_t)c * FINc + kk);
      acc[dt] = __builtin_amdgcn_wmma_f32_16x16x4_f32(
          false, a, false, bf, (short)0, acc[dt], false, false);
    }
  }

  // D layout: VGPR r -> (M = r + 8*half, N = lane%16)
#pragma unroll
  for (int dt = 0; dt < 4; ++dt) {
    const int d = dt * 16 + iLoc;
    _Float16* trow = hpT16 + ((size_t)(b * Hc + head) * Dc + d) * Nc;
#pragma unroll
    for (int r = 0; r < 8; ++r) {
      const int n = n0 + r + 8 * half;
      hp[(((size_t)b * Hc + head) * Nc + n) * Dc + d] = acc[dt][r];
    }
    // packed f16 pairs along n (consecutive D-fragment VGPRs)
#pragma unroll
    for (int r = 0; r < 8; r += 2) {
      v2h p; p.x = (_Float16)acc[dt][r]; p.y = (_Float16)acc[dt][r + 1];
      *(v2h*)(trow + n0 + 8 * half + r) = p;
    }
  }
}

// ---------------------------------------------------------------------------
// Kernel 2: s[b,h,n] = hp . a_src[h],  t likewise with a_dst.
// ---------------------------------------------------------------------------
__global__ __launch_bounds__(256) void k_st(const float* __restrict__ hp,
                                            const float* __restrict__ attn,
                                            float* __restrict__ sArr,
                                            float* __restrict__ tArr) {
  const int idx = blockIdx.x * 256 + threadIdx.x;   // (b*H + h)*N + n
  const int head = (idx / Nc) & (Hc - 1);
  const float* row  = hp + (size_t)idx * Dc;
  const float* aSrc = attn + (size_t)head * 2 * Dc;
  const float* aDst = aSrc + Dc;
  float s = 0.0f, t = 0.0f;
#pragma unroll 8
  for (int d = 0; d < Dc; ++d) {
    const float v = row[d];
    s = fmaf(v, aSrc[d], s);
    t = fmaf(v, aDst[d], t);
  }
  sArr[idx] = s;
  tArr[idx] = t;
}

// ---------------------------------------------------------------------------
// Kernel 3: per (b,i): masked-softmax stats (rowmax m, 1/sum il), all heads.
// One wave32 per row; adj read once; __shfl_xor tree combine.
// ---------------------------------------------------------------------------
__global__ __launch_bounds__(128) void k_stats(const int* __restrict__ adj,
                                               const float* __restrict__ sArr,
                                               const float* __restrict__ tArr,
                                               float* __restrict__ mArr,
                                               float* __restrict__ ilArr) {
  const int lane = threadIdx.x & 31;
  const int wave = threadIdx.x >> 5;
  const int row  = blockIdx.x * 4 + wave;           // [0, B*N)
  const int b = row >> 11;
  const int i = row & (Nc - 1);
  const float NEG_INF = -__builtin_inff();

  float sH[Hc], mH[Hc], lH[Hc];
#pragma unroll
  for (int hh = 0; hh < Hc; ++hh) {
    sH[hh] = sArr[((size_t)b * Hc + hh) * Nc + i];
    mH[hh] = NEG_INF;
    lH[hh] = 0.0f;
  }

  const int* adjRow = adj + ((size_t)b * Nc + i) * Nc;
  for (int j = lane; j < Nc; j += 32) {
    const int a = adjRow[j];
    if (a != 0) {
#pragma unroll
      for (int hh = 0; hh < Hc; ++hh) {
        const float t = tArr[((size_t)b * Hc + hh) * Nc + j];
        const float e = lrelu(sH[hh] + t);
        const float mOld = mH[hh];
        const float mNew = fmaxf(mOld, e);
        lH[hh] = lH[hh] * __expf(mOld - mNew) + __expf(e - mNew);
        mH[hh] = mNew;
      }
    }
  }

#pragma unroll
  for (int off = 16; off > 0; off >>= 1) {
#pragma unroll
    for (int hh = 0; hh < Hc; ++hh) {
      const float mO = __shfl_xor(mH[hh], off, 32);
      const float lO = __shfl_xor(lH[hh], off, 32);
      const float mN = fmaxf(mH[hh], mO);
      const float la = (mH[hh] > NEG_INF) ? lH[hh] * __expf(mH[hh] - mN) : 0.0f;
      const float lb = (mO    > NEG_INF) ? lO    * __expf(mO    - mN) : 0.0f;
      mH[hh] = mN;
      lH[hh] = la + lb;
    }
  }

  if (lane == 0) {
#pragma unroll
    for (int hh = 0; hh < Hc; ++hh) {
      const size_t o = ((size_t)b * Hc + hh) * Nc + i;
      mArr[o]  = mH[hh];
      ilArr[o] = 1.0f / lH[hh];
    }
  }
}

// ---------------------------------------------------------------------------
// Kernel 4: h_new = alpha @ hp (V_WMMA_F32_16X16X32_F16) + alpha_mean (fp32).
// One block (4 waves) per (b, 16-row i-tile); wave w owns j in [w*512,(w+1)*512)
// for ALL heads. t[b] (32KB) is staged into LDS via global_load_async_to_lds.
// Alpha A-fragments (16x32 f16) built in-register in the ISA VGPR layout:
//   lane<16: elems 0..7 -> K=0..7, elems 8..15 -> K=16..23
//   lane>=16: elems 0..7 -> K=8..15, elems 8..15 -> K=24..31
// B-fragments: two b128 loads/lane from the transposed f16 copy hpT16.
// Cross-wave accumulator reduction via LDS ds_add_f32.
// ---------------------------------------------------------------------------
__global__ __launch_bounds__(128) void k_attn(const int* __restrict__ adj,
                                              const _Float16* __restrict__ hpT16,
                                              const float* __restrict__ sArr,
                                              const float* __restrict__ tArr,
                                              const float* __restrict__ mArr,
                                              const float* __restrict__ ilArr,
                                              float* __restrict__ out) {
  __shared__ float tLds[Hc * Nc];                   // 32 KB: t for this batch
  __shared__ float accLds[16 * FOUTc];              // 16 KB: output reduction

  const int lane = threadIdx.x & 31;
  const int wave = threadIdx.x >> 5;
  const int iLoc = lane & 15;
  const int half = lane >> 4;
  const int tile = blockIdx.x;                       // [0, B*N/16)
  const int b    = tile / (Nc / 16);
  const int i0   = (tile % (Nc / 16)) * 16;
  const int i    = i0 + iLoc;

  // ---- async LDS fill of t[b][h][j] (contiguous 8192 floats) ----
  {
    const float* tSrc = tArr + (size_t)b * Hc * Nc;
    for (int x = threadIdx.x * 4; x < Hc * Nc; x += 128 * 4) {
      const unsigned ldsAddr = (unsigned)(__UINTPTR_TYPE__)(lds_f*)&tLds[x];
      const float* g = tSrc + x;
      asm volatile("global_load_async_to_lds_b128 %0, %1, off"
                   :: "v"(ldsAddr), "v"(g) : "memory");
    }
    asm volatile("s_wait_asynccnt 0x0" ::: "memory");
  }
  for (int x = threadIdx.x; x < 16 * FOUTc; x += 128) accLds[x] = 0.0f;
  __syncthreads();

  float sH[Hc], mH[Hc], ilH[Hc];
#pragma unroll
  for (int hh = 0; hh < Hc; ++hh) {
    const size_t o = ((size_t)b * Hc + hh) * Nc + i;
    sH[hh]  = sArr[o];
    mH[hh]  = mArr[o];
    ilH[hh] = ilArr[o];
  }

  const int jBeg = wave * (Nc / 4);
  const int jEnd = jBeg + (Nc / 4);
  const int kOff = half * 8;                         // A-fragment K base
  const int* adjRow = adj + ((size_t)b * Nc + i) * Nc;
  float* alphaOut = out + (size_t)Bc * Nc * FOUTc + ((size_t)b * Nc + i) * Nc;

  v8f acc[Hc][4] = {};                               // 4 heads x 4 d-tiles

  for (int jb = jBeg; jb < jEnd; jb += 32) {
    const int jA = jb + kOff;                        // lane's first j run
    __builtin_prefetch(adjRow + jb + 256, 0, 1);     // global_prefetch_b8

    // adjacency for this lane's 16 (i,j) pairs: runs [jA,jA+8) and [jA+16,jA+24)
    const int4 q0 = *(const int4*)(adjRow + jA);
    const int4 q1 = *(const int4*)(adjRow + jA + 4);
    const int4 q2 = *(const int4*)(adjRow + jA + 16);
    const int4 q3 = *(const int4*)(adjRow + jA + 20);
    int adjV[16] = {q0.x,q0.y,q0.z,q0.w, q1.x,q1.y,q1.z,q1.w,
                    q2.x,q2.y,q2.z,q2.w, q3.x,q3.y,q3.z,q3.w};

    float amv[16];
#pragma unroll
    for (int e = 0; e < 16; ++e) amv[e] = 0.0f;

#pragma unroll
    for (int hh = 0; hh < Hc; ++hh) {
      // t from LDS (ds_load_b128 x4)
      const float4 t0 = *(const float4*)&tLds[hh * Nc + jA];
      const float4 t1 = *(const float4*)&tLds[hh * Nc + jA + 4];
      const float4 t2 = *(const float4*)&tLds[hh * Nc + jA + 16];
      const float4 t3 = *(const float4*)&tLds[hh * Nc + jA + 20];
      float tV[16] = {t0.x,t0.y,t0.z,t0.w, t1.x,t1.y,t1.z,t1.w,
                      t2.x,t2.y,t2.z,t2.w, t3.x,t3.y,t3.z,t3.w};

      v16h aF;
#pragma unroll
      for (int e = 0; e < 16; ++e) {
        const float ev = lrelu(sH[hh] + tV[e]);
        const float al = adjV[e] ? __expf(ev - mH[hh]) * ilH[hh] : 0.0f;
        amv[e] += al;
        aF[e] = (_Float16)al;
      }

#pragma unroll
      for (int dt = 0; dt < 4; ++dt) {
        const int d = dt * 16 + iLoc;
        const _Float16* bp =
            hpT16 + ((size_t)(b * Hc + hh) * Dc + d) * Nc + jb + half * 8;
        const v8h blo = *(const v8h*)bp;             // K = base..base+7
        const v8h bhi = *(const v8h*)(bp + 16);      // K = base+16..base+23
        const v16h bF = __builtin_shufflevector(blo, bhi,
            0,1,2,3,4,5,6,7,8,9,10,11,12,13,14,15);
        acc[hh][dt] = __builtin_amdgcn_wmma_f32_16x16x32_f16(
            false, aF, false, bF, (short)0, acc[hh][dt], false, false);
      }
    }

    // exclusive alpha_mean region: (i, jA..jA+7) and (i, jA+16..jA+23)
    float4 s0, s1, s2, s3;
    s0.x=amv[0]*0.25f;  s0.y=amv[1]*0.25f;  s0.z=amv[2]*0.25f;  s0.w=amv[3]*0.25f;
    s1.x=amv[4]*0.25f;  s1.y=amv[5]*0.25f;  s1.z=amv[6]*0.25f;  s1.w=amv[7]*0.25f;
    s2.x=amv[8]*0.25f;  s2.y=amv[9]*0.25f;  s2.z=amv[10]*0.25f; s2.w=amv[11]*0.25f;
    s3.x=amv[12]*0.25f; s3.y=amv[13]*0.25f; s3.z=amv[14]*0.25f; s3.w=amv[15]*0.25f;
    *(float4*)(alphaOut + jA)      = s0;
    *(float4*)(alphaOut + jA + 4)  = s1;
    *(float4*)(alphaOut + jA + 16) = s2;
    *(float4*)(alphaOut + jA + 20) = s3;
  }

  // Reduce the 4 waves' partial accumulators via LDS float atomics.
#pragma unroll
  for (int hh = 0; hh < Hc; ++hh)
#pragma unroll
    for (int dt = 0; dt < 4; ++dt)
#pragma unroll
      for (int r = 0; r < 8; ++r) {
        const int rowL = r + 8 * half;               // D-matrix M index
        const int col  = hh * Dc + dt * 16 + iLoc;
        atomicAdd(&accLds[rowL * FOUTc + col], acc[hh][dt][r]);
      }
  __syncthreads();

  // h_new[b, i0+r, :] — coalesced block store
  for (int x = threadIdx.x; x < 16 * FOUTc; x += 128) {
    const int r = x >> 8;
    const int c = x & (FOUTc - 1);
    out[((size_t)b * Nc + i0 + r) * FOUTc + c] = accLds[x];
  }
}

// ---------------------------------------------------------------------------
extern "C" void kernel_launch(void* const* d_in, const int* in_sizes, int n_in,
                              void* d_out, int out_size, void* d_ws, size_t ws_size,
                              hipStream_t stream) {
  const float* h    = (const float*)d_in[0];
  const int*   adj  = (const int*)d_in[1];
  const float* W    = (const float*)d_in[2];
  const float* attn = (const float*)d_in[3];
  float* out = (float*)d_out;

  const size_t BHN = (size_t)Bc * Hc * Nc;
  float* hp     = (float*)d_ws;                      // B*H*N*D fp32
  float* sArr   = hp   + BHN * Dc;
  float* tArr   = sArr + BHN;
  float* mArr   = tArr + BHN;
  float* ilArr  = mArr + BHN;
  _Float16* hpT16 = (_Float16*)(ilArr + BHN);        // B*H*D*N f16 (transposed)

  // 1) projection GEMM (fp32 WMMA) + transposed f16 copy
  k_project<<<dim3(Bc * Nc / 16, Hc), 32, 0, stream>>>(h, W, hp, hpT16);

  // 2) s,t per (b,h,n)
  k_st<<<(Bc * Hc * Nc) / 256, 256, 0, stream>>>(hp, attn, sArr, tArr);

  // 3) softmax stats per (b,i) row
  k_stats<<<(Bc * Nc) / 4, 128, 0, stream>>>(adj, sArr, tArr, mArr, ilArr);

  // 4) PV GEMM (f16 WMMA, K=32) + alpha_mean
  k_attn<<<(Bc * Nc) / 16, 128, 0, stream>>>(adj, hpT16, sArr, tArr, mArr, ilArr, out);
}
